// HypergraphDrugEncoder_71717363908781
// MI455X (gfx1250) — matrix operand order repaired and checked by reading, not compiled, not run on hardware
//
#include <hip/hip_runtime.h>
#include <hip/hip_bf16.h>
#include <math.h>

// ---------------- problem constants (match reference) ----------------
#define N_NODES   100000
#define N_HEDGES  50000
#define N_INC     300000
#define N_GRAPHS  1024
#define D_IN      49
#define DM        512
#define NLAYERS   3
#define LN_EPS    1e-5f

typedef __attribute__((ext_vector_type(16))) __bf16 v16bf;
typedef __attribute__((ext_vector_type(8)))  __bf16 v8bf;
typedef __attribute__((ext_vector_type(8)))  float  v8f;

#define BCOLS   256            // columns per block
#define KSTEP   32             // K per WMMA
#define BPAD    40             // padded K-stride (bf16 elems) per column in LDS

// =====================================================================
// WMMA GEMM:  Out[m][n] = (rs[m] * A[m][:]) dot Wt[n][:] + bias[n]
// A: f32 row-major [M x 512]; Wt: bf16, Wt[n*512+k] (transposed W)
// Block: 256 threads = 8 waves; block tile 128 rows x 256 cols.
// B slabs (32K x 256N, 16KB) double-buffered through LDS, shared by 8 waves.
// Inner column-tile loop software-pipelined 3 deep on B fragments.
// Grid: (ceil(M/128), DM/256)
// =====================================================================
__global__ __launch_bounds__(256)
void wmma_gemm_kernel(const float* __restrict__ A,
                      const __bf16* __restrict__ Wt,
                      const float* __restrict__ bias,
                      const float* __restrict__ row_scale,
                      float* __restrict__ Out,
                      int mtiles)               // M / 16
{
    __shared__ __align__(32) __bf16 Bsh[2][BCOLS * BPAD];   // 2 x 20KB

    const int tid   = threadIdx.x;
    const int wave  = tid >> 5;
    const int lane  = tid & 31;
    const int ml    = lane & 15;
    const int half  = lane >> 4;
    const int rowtile = blockIdx.x * 8 + wave;         // 16-row tile index
    const bool active = rowtile < mtiles;
    const int n0    = blockIdx.y * BCOLS;

    const int row   = rowtile * 16 + ml;               // A row this lane loads
    const float rs  = (active && row_scale) ? row_scale[row] : 1.0f;
    const float* arow = active ? (A + (size_t)row * DM) : A;

    v8f acc[16];
#pragma unroll
    for (int t = 0; t < 16; ++t) acc[t] = (v8f)(0.0f);

    // ---- stage slab kt=0 into buffer 0 (each thread copies one column) ----
    {
        const __bf16* src = Wt + (((size_t)(n0 + tid)) << 9);
        __bf16* dst = &Bsh[0][tid * BPAD];
#pragma unroll
        for (int c = 0; c < 4; ++c)
            *(v8bf*)(dst + c * 8) = *(const v8bf*)(src + c * 8);
    }
    __syncthreads();

    int buf = 0;
#pragma unroll 1
    for (int kt = 0; kt < 16; ++kt) {
        const int kb = kt * KSTEP;

        // ---- stage next slab into the other buffer ----
        if (kt < 15) {
            const __bf16* src = Wt + (((size_t)(n0 + tid)) << 9) + kb + KSTEP;
            __bf16* dst = &Bsh[buf ^ 1][tid * BPAD];
#pragma unroll
            for (int c = 0; c < 4; ++c)
                *(v8bf*)(dst + c * 8) = *(const v8bf*)(src + c * 8);
        }

        if (active) {
            // A fragment: element j -> K = kb + half*8 + (j<8 ? j : j+8)
            const float* a0 = arow + kb + half * 8;
            const float* a1 = a0 + 16;
            v16bf af;
#pragma unroll
            for (int j = 0; j < 8; ++j) {
                af[j]     = (__bf16)(a0[j] * rs);
                af[j + 8] = (__bf16)(a1[j] * rs);
            }

            // 16 column tiles; B fragments from LDS, pipelined 3 deep
            const __bf16* bbase = &Bsh[buf][ml * BPAD + half * 8];
            union U { v16bf v; v8bf h[2]; };
            U b[3];
#pragma unroll
            for (int i = 0; i < 3; ++i) {
                b[i].h[0] = *(const v8bf*)(bbase + i * 16 * BPAD);
                b[i].h[1] = *(const v8bf*)(bbase + i * 16 * BPAD + 16);
            }
#pragma unroll
            for (int t = 0; t < 16; ++t) {
                acc[t] = __builtin_amdgcn_wmma_f32_16x16x32_bf16(
                             false, af, false, b[t % 3].v, (short)0, acc[t], false, false);
                if (t + 3 < 16) {
                    b[t % 3].h[0] = *(const v8bf*)(bbase + (t + 3) * 16 * BPAD);
                    b[t % 3].h[1] = *(const v8bf*)(bbase + (t + 3) * 16 * BPAD + 16);
                }
            }
        }
        __syncthreads();
        buf ^= 1;
    }

    if (active) {
        // D layout: lane -> N = ml; element r -> M = half*8 + r
#pragma unroll
        for (int t = 0; t < 16; ++t) {
            const int n = n0 + t * 16 + ml;
            const float bn = bias[n];
#pragma unroll
            for (int r = 0; r < 8; ++r) {
                const size_t m = (size_t)rowtile * 16 + half * 8 + r;
                Out[(m << 9) + n] = acc[t][r] + bn;
            }
        }
    }
}

// ---------------- weight convert + transpose to bf16 ----------------
__global__ void wt_cvt_kernel(const float* __restrict__ w, __bf16* __restrict__ wt, int total)
{
    int i = blockIdx.x * blockDim.x + threadIdx.x;
    if (i >= total) return;
    int mat = i >> 18;                 // / (512*512)
    int rem = i & (262144 - 1);
    int k = rem >> 9, n = rem & 511;
    wt[(size_t)mat * 262144 + ((size_t)n << 9) + k] = (__bf16)w[(size_t)mat * 262144 + ((size_t)k << 9) + n];
}

// ---------------- degree / count kernels ----------------
__global__ void deg_kernel(const int* __restrict__ nidx, const int* __restrict__ hidx,
                           float* __restrict__ ncnt, float* __restrict__ hcnt, int n)
{
    int e = blockIdx.x * blockDim.x + threadIdx.x;
    if (e >= n) return;
    atomicAdd(&ncnt[nidx[e]], 1.0f);
    atomicAdd(&hcnt[hidx[e]], 1.0f);
}

__global__ void batch_cnt_kernel(const int* __restrict__ b, float* __restrict__ cnt, int n)
{
    int i = blockIdx.x * blockDim.x + threadIdx.x;
    if (i >= n) return;
    atomicAdd(&cnt[b[i]], 1.0f);
}

__global__ void inv_clip_kernel(float* __restrict__ a, int n)
{
    int i = blockIdx.x * blockDim.x + threadIdx.x;
    if (i >= n) return;
    a[i] = 1.0f / fmaxf(a[i], 1.0f);
}

// ---------------- input projection: x = feat @ in_w + in_b ----------------
__global__ __launch_bounds__(256)
void in_gemm_kernel(const float* __restrict__ feat, const float* __restrict__ w,
                    const float* __restrict__ b, float* __restrict__ x)
{
    const int m = blockIdx.x;
    const int n = threadIdx.x;
    const float* f = feat + (size_t)m * D_IN;
    float acc0 = b[n], acc1 = b[n + 256];
#pragma unroll
    for (int k = 0; k < D_IN; ++k) {
        const float fv = f[k];
        acc0 += fv * w[(size_t)k * DM + n];
        acc1 += fv * w[(size_t)k * DM + n + 256];
    }
    x[((size_t)m << 9) + n]       = acc0;
    x[((size_t)m << 9) + n + 256] = acc1;
}

// ---------------- scatter: incidences -> hedge buffer ----------------
__global__ void hedge_scatter_kernel(const float* __restrict__ xt,
                                     const int* __restrict__ nidx, const int* __restrict__ hidx,
                                     float* __restrict__ hbuf)
{
    int t = blockIdx.x * blockDim.x + threadIdx.x;     // N_INC * 128 threads
    int e = t >> 7;
    if (e >= N_INC) return;
    int d = (t & 127) << 2;
    const float4 v = *(const float4*)(xt + ((size_t)nidx[e] << 9) + d);
    float* dst = hbuf + ((size_t)hidx[e] << 9) + d;
    atomicAdd(dst + 0, v.x); atomicAdd(dst + 1, v.y);
    atomicAdd(dst + 2, v.z); atomicAdd(dst + 3, v.w);
}

// hedge_msg = hedge_aggr/deg + type_emb[type]
__global__ void hedge_fin_kernel(float* __restrict__ hbuf, const float* __restrict__ hinv,
                                 const int* __restrict__ types, const float* __restrict__ temb)
{
    int i = blockIdx.x * blockDim.x + threadIdx.x;     // N_HEDGES*512
    if (i >= N_HEDGES * DM) return;
    int h = i >> 9, d = i & 511;
    hbuf[i] = hbuf[i] * hinv[h] + temb[(size_t)types[h] * DM + d];
}

// ---------------- scatter: hedges -> node aggregate ----------------
__global__ void node_scatter_kernel(const float* __restrict__ hbuf,
                                    const int* __restrict__ nidx, const int* __restrict__ hidx,
                                    float* __restrict__ nagg)
{
    int t = blockIdx.x * blockDim.x + threadIdx.x;
    int e = t >> 7;
    if (e >= N_INC) return;
    int d = (t & 127) << 2;
    const float4 v = *(const float4*)(hbuf + ((size_t)hidx[e] << 9) + d);
    float* dst = nagg + ((size_t)nidx[e] << 9) + d;
    atomicAdd(dst + 0, v.x); atomicAdd(dst + 1, v.y);
    atomicAdd(dst + 2, v.z); atomicAdd(dst + 3, v.w);
}

// ---------------- fused residual + LayerNorm + exact GELU ----------------
__global__ __launch_bounds__(256)
void ln_gelu_kernel(const float* __restrict__ y, float* __restrict__ x,
                    const float* __restrict__ g, const float* __restrict__ b)
{
    __shared__ float r1[256], r2[256];
    const size_t base = ((size_t)blockIdx.x) << 9;
    const int tid = threadIdx.x;
    float v0 = y[base + tid]       + x[base + tid];
    float v1 = y[base + tid + 256] + x[base + tid + 256];
    r1[tid] = v0 + v1;
    r2[tid] = v0 * v0 + v1 * v1;
    __syncthreads();
    for (int o = 128; o > 0; o >>= 1) {
        if (tid < o) { r1[tid] += r1[tid + o]; r2[tid] += r2[tid + o]; }
        __syncthreads();
    }
    const float mu   = r1[0] * (1.0f / DM);
    const float var  = r2[0] * (1.0f / DM) - mu * mu;
    const float rstd = rsqrtf(var + LN_EPS);
    float u0 = (v0 - mu) * rstd * g[tid]       + b[tid];
    float u1 = (v1 - mu) * rstd * g[tid + 256] + b[tid + 256];
    x[base + tid]       = 0.5f * u0 * (1.0f + erff(u0 * 0.70710678118654752f));
    x[base + tid + 256] = 0.5f * u1 * (1.0f + erff(u1 * 0.70710678118654752f));
}

// ---------------- graph pooling scatter ----------------
__global__ void graph_scatter_kernel(const float* __restrict__ x, const int* __restrict__ batch,
                                     float* __restrict__ gsum)
{
    int t = blockIdx.x * blockDim.x + threadIdx.x;     // N_NODES * 128
    int m = t >> 7;
    if (m >= N_NODES) return;
    int d = (t & 127) << 2;
    const float4 v = *(const float4*)(x + ((size_t)m << 9) + d);
    float* dst = gsum + ((size_t)batch[m] << 9) + d;
    atomicAdd(dst + 0, v.x); atomicAdd(dst + 1, v.y);
    atomicAdd(dst + 2, v.z); atomicAdd(dst + 3, v.w);
}

__global__ void batch_to_f32_kernel(const int* __restrict__ b, float* __restrict__ o, int n)
{
    int i = blockIdx.x * blockDim.x + threadIdx.x;
    if (i < n) o[i] = (float)b[i];
}

// =====================================================================
extern "C" void kernel_launch(void* const* d_in, const int* in_sizes, int n_in,
                              void* d_out, int out_size, void* d_ws, size_t ws_size,
                              hipStream_t stream)
{
    const float* node_features = (const float*)d_in[0];
    const int*   node_idx      = (const int*)  d_in[1];
    const int*   hedge_idx     = (const int*)  d_in[2];
    const int*   hedge_types   = (const int*)  d_in[3];
    const int*   batch_idx     = (const int*)  d_in[4];
    const float* in_w          = (const float*)d_in[6];
    const float* in_b          = (const float*)d_in[7];
    const float* type_emb      = (const float*)d_in[8];
    const float* l1_w          = (const float*)d_in[9];
    const float* l1_b          = (const float*)d_in[10];
    const float* l2_w          = (const float*)d_in[11];
    const float* l2_b          = (const float*)d_in[12];
    const float* ln_g          = (const float*)d_in[13];
    const float* ln_b          = (const float*)d_in[14];
    const float* out_w         = (const float*)d_in[15];
    const float* out_b         = (const float*)d_in[16];

    // ---------------- workspace layout ----------------
    char* base = (char*)d_ws;
    const size_t XN = (size_t)N_NODES * DM;            // 51,200,000
    float*  x     = (float*)(base);                           // 204.8 MB
    float*  xt    = (float*)(base + XN * 4);                  // 204.8 MB (x_trans / y)
    float*  nagg  = (float*)(base + 2 * XN * 4);              // 204.8 MB
    float*  hbuf  = (float*)(base + 3 * XN * 4);              // 102.4 MB
    char*   p     = base + 3 * XN * 4 + (size_t)N_HEDGES * DM * 4;
    float*  gsum  = (float*)p;            p += (size_t)N_GRAPHS * DM * 4;
    float*  ninv  = (float*)p;            p += (size_t)N_NODES * 4;
    float*  hinv  = (float*)p;            p += (size_t)N_HEDGES * 4;
    float*  ginv  = (float*)p;            p += (size_t)N_GRAPHS * 4;
    __bf16* l1wt  = (__bf16*)p;           p += (size_t)NLAYERS * DM * DM * 2;
    __bf16* l2wt  = (__bf16*)p;           p += (size_t)NLAYERS * DM * DM * 2;
    __bf16* outwt = (__bf16*)p;           p += (size_t)DM * DM * 2;

    float* out_graph = (float*)d_out;                         // [1024 x 512]
    float* out_nodes = out_graph + (size_t)N_GRAPHS * DM;     // [100000 x 512]
    float* out_batch = out_nodes + XN;                        // [100000]

    // ---------------- degrees (loop-invariant) ----------------
    hipMemsetAsync(ninv, 0, (size_t)N_NODES * 4, stream);
    hipMemsetAsync(hinv, 0, (size_t)N_HEDGES * 4, stream);
    hipMemsetAsync(ginv, 0, (size_t)N_GRAPHS * 4, stream);
    deg_kernel<<<(N_INC + 255) / 256, 256, 0, stream>>>(node_idx, hedge_idx, ninv, hinv, N_INC);
    batch_cnt_kernel<<<(N_NODES + 255) / 256, 256, 0, stream>>>(batch_idx, ginv, N_NODES);
    inv_clip_kernel<<<(N_NODES + 255) / 256, 256, 0, stream>>>(ninv, N_NODES);
    inv_clip_kernel<<<(N_HEDGES + 255) / 256, 256, 0, stream>>>(hinv, N_HEDGES);
    inv_clip_kernel<<<(N_GRAPHS + 255) / 256, 256, 0, stream>>>(ginv, N_GRAPHS);

    // ---------------- weight convert/transpose to bf16 ----------------
    {
        const int tl = NLAYERS * DM * DM;
        wt_cvt_kernel<<<(tl + 255) / 256, 256, 0, stream>>>(l1_w, l1wt, tl);
        wt_cvt_kernel<<<(tl + 255) / 256, 256, 0, stream>>>(l2_w, l2wt, tl);
        const int t1 = DM * DM;
        wt_cvt_kernel<<<(t1 + 255) / 256, 256, 0, stream>>>(out_w, outwt, t1);
    }

    // ---------------- input projection ----------------
    in_gemm_kernel<<<N_NODES, 256, 0, stream>>>(node_features, in_w, in_b, x);

    // ---------------- layers ----------------
    const int mtiles = N_NODES / 16;                       // 6250
    const dim3 ggrid((mtiles + 7) / 8, DM / BCOLS);        // 782 x 2
    for (int i = 0; i < NLAYERS; ++i) {
        // x_trans = x @ l1_w[i] + l1_b[i]
        wmma_gemm_kernel<<<ggrid, 256, 0, stream>>>(x, l1wt + (size_t)i * DM * DM,
                                                    l1_b + (size_t)i * DM, nullptr, xt, mtiles);
        // hedge aggregate + message
        hipMemsetAsync(hbuf, 0, (size_t)N_HEDGES * DM * 4, stream);
        hedge_scatter_kernel<<<(N_INC * 128) / 256, 256, 0, stream>>>(xt, node_idx, hedge_idx, hbuf);
        hedge_fin_kernel<<<(N_HEDGES * DM) / 256, 256, 0, stream>>>(hbuf, hinv, hedge_types, type_emb);
        // node aggregate
        hipMemsetAsync(nagg, 0, XN * 4, stream);
        node_scatter_kernel<<<(N_INC * 128) / 256, 256, 0, stream>>>(hbuf, node_idx, hedge_idx, nagg);
        // y = (nagg/node_deg) @ l2_w[i] + l2_b[i]   (degree folded into A convert)
        wmma_gemm_kernel<<<ggrid, 256, 0, stream>>>(nagg, l2wt + (size_t)i * DM * DM,
                                                    l2_b + (size_t)i * DM, ninv, xt, mtiles);
        // x = gelu(LN(x + y))
        ln_gelu_kernel<<<N_NODES, 256, 0, stream>>>(xt, x, ln_g + (size_t)i * DM, ln_b + (size_t)i * DM);
    }

    // ---------------- graph pooling + readout ----------------
    hipMemsetAsync(gsum, 0, (size_t)N_GRAPHS * DM * 4, stream);
    graph_scatter_kernel<<<(N_NODES * 128) / 256, 256, 0, stream>>>(x, batch_idx, gsum);
    {
        const int gmt = N_GRAPHS / 16;                     // 64
        wmma_gemm_kernel<<<dim3((gmt + 7) / 8, DM / BCOLS), 256, 0, stream>>>(
            gsum, outwt, out_b, ginv, out_graph, gmt);
    }

    // ---------------- outputs ----------------
    hipMemcpyAsync(out_nodes, x, XN * 4, hipMemcpyDeviceToDevice, stream);
    batch_to_f32_kernel<<<(N_NODES + 255) / 256, 256, 0, stream>>>(batch_idx, out_batch, N_NODES);
}
